// Attention_78005196030545
// MI455X (gfx1250) — matrix operand order repaired and checked by reading, not compiled, not run on hardware
//
#include <hip/hip_runtime.h>

// Fused eager-attention for B=2,S=2048,H=16,DH=64,D=1024 on gfx1250.
// Mixed precision: f16 WMMA (v_wmma_f32_16x16x32_f16) with fp32 accumulate,
// fp32 softmax. Weights/activations pre-converted to f16 (weights transposed)
// so every WMMA operand is a contiguous 16/32-byte vector load.
// Intermediates: 48 MB workspace, resident in the 192 MB L2.

#define B_  2
#define S_  2048
#define H_  16
#define DH_ 64
#define D_  1024
#define SCALE_ 0.125f   // 1/sqrt(64)

typedef _Float16 h16;
typedef __attribute__((ext_vector_type(16))) _Float16 v16h;
typedef __attribute__((ext_vector_type(8)))  _Float16 v8h;
typedef __attribute__((ext_vector_type(8)))  float    v8f;

static __device__ __forceinline__ v8f wmma_f16(v16h a, v16h b, v8f c) {
  return __builtin_amdgcn_wmma_f32_16x16x32_f16(
      /*neg_a=*/false, a, /*neg_b=*/false, b,
      /*c_mod=*/(short)0, c, /*reuse_a=*/false, /*reuse_b=*/false);
}

// A-fragment (16 rows x 32 contraction) from a row-major f16 matrix.
// `row` points at this lane's matrix row (lane&15), `dc` = contraction base.
// Layout: lanes 0-15: elems 0..15 = k dc+0..15
//         lanes16-31: elems 0..15 = k dc+16..31
static __device__ __forceinline__ v16h load_afrag(const h16* __restrict__ row,
                                                  int dc, int half_) {
  v8h lo = *(const v8h*)(row + dc + 16 * half_);
  v8h hi = *(const v8h*)(row + dc + 16 * half_ + 8);
  v16h a;
#pragma unroll
  for (int j = 0; j < 8; ++j) { a[j] = lo[j]; a[8 + j] = hi[j]; }
  return a;
}

// ---------------------------------------------------------------------------
// Kernel 0a: X (fp32) -> Xh (f16), elementwise, 8 elems/thread.
// ---------------------------------------------------------------------------
__global__ __launch_bounds__(256) void xcvt_kernel(const float* __restrict__ X,
                                                   h16* __restrict__ Xh) {
  const size_t i = ((size_t)blockIdx.x * 256 + threadIdx.x) * 8;
  float4 a = *(const float4*)(X + i);
  float4 b = *(const float4*)(X + i + 4);
  v8h o;
  o[0] = (h16)a.x; o[1] = (h16)a.y; o[2] = (h16)a.z; o[3] = (h16)a.w;
  o[4] = (h16)b.x; o[5] = (h16)b.y; o[6] = (h16)b.z; o[7] = (h16)b.w;
  *(v8h*)(Xh + i) = o;
}

// ---------------------------------------------------------------------------
// Kernel 0b: W (fp32, [k][n]) -> Wt (f16, [n][k]) via LDS 32x32 tile transpose.
// grid.z selects which of the 4 weight matrices.
// ---------------------------------------------------------------------------
__global__ __launch_bounds__(256) void wcvt_kernel(
    const float* __restrict__ Wq, const float* __restrict__ Wk,
    const float* __restrict__ Wv, const float* __restrict__ Wo,
    h16* __restrict__ WtQ, h16* __restrict__ WtK,
    h16* __restrict__ WtV, h16* __restrict__ WtO) {
  const int m = blockIdx.z;
  const float* W = (m == 0) ? Wq : (m == 1) ? Wk : (m == 2) ? Wv : Wo;
  h16* Wt       = (m == 0) ? WtQ : (m == 1) ? WtK : (m == 2) ? WtV : WtO;
  __shared__ h16 tile[32][33];
  const int tx = threadIdx.x & 31, ty = threadIdx.x >> 5;
  const int k0 = blockIdx.x * 32, n0 = blockIdx.y * 32;
#pragma unroll
  for (int r = ty; r < 32; r += 8)
    tile[r][tx] = (h16)W[(size_t)(k0 + r) * D_ + n0 + tx];
  __syncthreads();
#pragma unroll
  for (int r = ty; r < 32; r += 8)
    Wt[(size_t)(n0 + r) * D_ + k0 + tx] = tile[tx][r];
}

// ---------------------------------------------------------------------------
// Kernel 1: fused Q/K/V projection.  grid.z selects the weight matrix.
// One wave per 32x64 output tile (2 A-frags x 4 B-frags -> 8 wmma per k-step,
// ~1 vector load per wmma).  64-col stripe == one head.
// Q,K -> [B,H,S,DH] f16.  V -> [B,H,DH,S] (transposed) so the PV GEMM's
// A-fragments are contiguous along keys.
// ---------------------------------------------------------------------------
__global__ __launch_bounds__(128) void qkv_proj_kernel(
    const h16* __restrict__ Xh,
    const h16* __restrict__ WtQ, const float* __restrict__ bq,
    const h16* __restrict__ WtK, const float* __restrict__ bk,
    const h16* __restrict__ WtV, const float* __restrict__ bv,
    h16* __restrict__ Qh, h16* __restrict__ Kh, h16* __restrict__ VhT) {
  const int mode = blockIdx.z;                    // 0=Q 1=K 2=V
  const h16*  Wt   = (mode == 0) ? WtQ : (mode == 1) ? WtK : WtV;
  const float* bias = (mode == 0) ? bq : (mode == 1) ? bk : bv;
  const int wave  = threadIdx.x >> 5;
  const int lane  = threadIdx.x & 31;
  const int half_ = lane >> 4;
  const int l16   = lane & 15;
  const int row0  = blockIdx.x * 32;                      // over B*S = 4096
  const int col0  = (blockIdx.y * 4 + wave) * 64;         // head stripe

  v8f acc[2][4] = {};
  const h16* ar0 = Xh + (size_t)(row0 + l16) * D_;
  const h16* ar1 = ar0 + (size_t)16 * D_;
  const h16* wr  = Wt + (size_t)(col0 + l16) * D_;        // + ct*16*D_ per col tile
  for (int kk = 0; kk < D_; kk += 32) {
    v16h a0 = load_afrag(ar0, kk, half_);
    v16h a1 = load_afrag(ar1, kk, half_);
#pragma unroll
    for (int ct = 0; ct < 4; ++ct) {
      v16h bf = *(const v16h*)(wr + (size_t)ct * 16 * D_ + kk + 16 * half_);
      acc[0][ct] = wmma_f16(a0, bf, acc[0][ct]);
      acc[1][ct] = wmma_f16(a1, bf, acc[1][ct]);
    }
  }

  const int h  = col0 >> 6;
  const int b  = row0 >> 11;          // 32 | 2048: tiles never straddle batch
#pragma unroll
  for (int ct = 0; ct < 4; ++ct) {
    const float bval = bias[col0 + ct * 16 + l16];
    const int dh = ct * 16 + l16;
#pragma unroll
    for (int rt = 0; rt < 2; ++rt) {
      const int s0 = (row0 & (S_ - 1)) + rt * 16 + 8 * half_;
      if (mode < 2) {
        h16* out = (mode == 0 ? Qh : Kh) + ((size_t)b * H_ + h) * S_ * DH_;
#pragma unroll
        for (int i = 0; i < 8; ++i)
          out[(size_t)(s0 + i) * DH_ + dh] = (h16)(acc[rt][ct][i] + bval);
      } else {                        // V transposed: contiguous along s
        v8h v;
#pragma unroll
        for (int i = 0; i < 8; ++i) v[i] = (h16)(acc[rt][ct][i] + bval);
        *(v8h*)(VhT + (((size_t)b * H_ + h) * DH_ + dh) * S_ + s0) = v;
      }
    }
  }
}

// ---------------------------------------------------------------------------
// Kernel 2: flash attention, one wave per (b, h, 16-query tile).
// Scores computed transposed (S^T = K * Q^T) so queries sit in lanes:
// per-query online softmax needs only 8 VALU ops + one shfl_xor(16).
// Context accumulated transposed (ctx^T = V^T * P^T); P^T B-fragment built
// from the exp'd score fragments with half-wave shuffles.
// ---------------------------------------------------------------------------
__global__ __launch_bounds__(256) void attn_kernel(
    const h16* __restrict__ Qh, const h16* __restrict__ Kh,
    const h16* __restrict__ VhT, h16* __restrict__ Ctx) {
  const int wave  = threadIdx.x >> 5;
  const int lane  = threadIdx.x & 31;
  const int half_ = lane >> 4;
  const int l16   = lane & 15;
  const int gw = blockIdx.x * 8 + wave;        // 4096 waves total
  const int qt = gw & 127;                     // S/16 query tiles
  const int h  = (gw >> 7) & (H_ - 1);
  const int b  = gw >> 11;
  const int q0 = qt * 16;

  const h16* Qbase = Qh  + ((size_t)b * H_ + h) * S_ * DH_;
  const h16* Kbase = Kh  + ((size_t)b * H_ + h) * S_ * DH_;
  const h16* Vbase = VhT + ((size_t)b * H_ + h) * DH_ * S_;

  // Q^T B-fragments, resident for the whole kv loop (d-chunks [0,32),[32,64))
  const h16* qrow = Qbase + (size_t)(q0 + l16) * DH_;
  v16h qb0 = *(const v16h*)(qrow + 16 * half_);
  v16h qb1 = *(const v16h*)(qrow + 32 + 16 * half_);

  float m = -3.0e38f, l = 0.f;
  v8f o0 = {}, o1 = {}, o2 = {}, o3 = {};     // ctx^T: 64 d-rows x 16 q-cols

  for (int k0 = 0; k0 < S_; k0 += 32) {
    // ---- scores^T for two 16-key tiles (rows=keys, cols=queries)
    v8f s[2];
#pragma unroll
    for (int t = 0; t < 2; ++t) {
      const h16* kr = Kbase + (size_t)(k0 + t * 16 + l16) * DH_;
      v8f sc = {};
      sc = wmma_f16(load_afrag(kr, 0, half_), qb0, sc);
      sc = wmma_f16(load_afrag(kr, 32, half_), qb1, sc);
#pragma unroll
      for (int i = 0; i < 8; ++i) sc[i] *= SCALE_;
      s[t] = sc;
    }
    // ---- online softmax (per query column = per lane, halves paired)
    float cm = s[0][0];
#pragma unroll
    for (int i = 0; i < 8; ++i) cm = fmaxf(cm, fmaxf(s[0][i], s[1][i]));
    cm = fmaxf(cm, __shfl_xor(cm, 16, 32));
    const float mn = fmaxf(m, cm);
    const float alpha = __expf(m - mn);
    m = mn;
    float rs = 0.f;
#pragma unroll
    for (int i = 0; i < 8; ++i) { s[0][i] = __expf(s[0][i] - mn); rs += s[0][i]; }
#pragma unroll
    for (int i = 0; i < 8; ++i) { s[1][i] = __expf(s[1][i] - mn); rs += s[1][i]; }
    rs += __shfl_xor(rs, 16, 32);
    l = l * alpha + rs;
#pragma unroll
    for (int i = 0; i < 8; ++i) {
      o0[i] *= alpha; o1[i] *= alpha; o2[i] *= alpha; o3[i] *= alpha;
    }
    // ---- P^T B-fragment (32 keys x 16 queries) via half-wave exchange
    v16h pb;
#pragma unroll
    for (int j = 0; j < 8; ++j) {
      const float xa = __shfl_xor(s[0][j], 16, 32);  // partner's key rows
      const float xb = __shfl_xor(s[1][j], 16, 32);
      pb[j]     = (h16)(half_ ? xb : s[0][j]);       // keys 0..7 / 16..23
      pb[8 + j] = (h16)(half_ ? s[1][j] : xa);       // keys 8..15 / 24..31
    }
    // ---- ctx^T += V^T(16d x 32k) * P^T(32k x 16q), 4 d-tiles
    o0 = wmma_f16(load_afrag(Vbase + (size_t)( 0 + l16) * S_ + k0, 0, half_), pb, o0);
    o1 = wmma_f16(load_afrag(Vbase + (size_t)(16 + l16) * S_ + k0, 0, half_), pb, o1);
    o2 = wmma_f16(load_afrag(Vbase + (size_t)(32 + l16) * S_ + k0, 0, half_), pb, o2);
    o3 = wmma_f16(load_afrag(Vbase + (size_t)(48 + l16) * S_ + k0, 0, half_), pb, o3);
  }

  // ---- normalize and store ctx as f16 [B,S,D] (row-major, heads interleaved)
  const float inv = 1.0f / l;
  h16* crow = Ctx + ((size_t)b * S_ + q0 + l16) * D_ + h * DH_;
  v8h t0, t1, t2, t3;
#pragma unroll
  for (int i = 0; i < 8; ++i) {
    t0[i] = (h16)(o0[i] * inv); t1[i] = (h16)(o1[i] * inv);
    t2[i] = (h16)(o2[i] * inv); t3[i] = (h16)(o3[i] * inv);
  }
  *(v8h*)(crow +  0 + 8 * half_) = t0;
  *(v8h*)(crow + 16 + 8 * half_) = t1;
  *(v8h*)(crow + 32 + 8 * half_) = t2;
  *(v8h*)(crow + 48 + 8 * half_) = t3;
}

// ---------------------------------------------------------------------------
// Kernel 3: output projection  out = Ctx(f16) @ Wo + bo -> fp32.
// Same 32x64 register tile as the QKV projection.
// ---------------------------------------------------------------------------
__global__ __launch_bounds__(128) void out_proj_kernel(
    const h16* __restrict__ Ctx, const h16* __restrict__ WtO,
    const float* __restrict__ bo, float* __restrict__ Out) {
  const int wave  = threadIdx.x >> 5;
  const int lane  = threadIdx.x & 31;
  const int half_ = lane >> 4;
  const int l16   = lane & 15;
  const int row0  = blockIdx.x * 32;
  const int col0  = (blockIdx.y * 4 + wave) * 64;

  v8f acc[2][4] = {};
  const h16* ar0 = Ctx + (size_t)(row0 + l16) * D_;
  const h16* ar1 = ar0 + (size_t)16 * D_;
  const h16* wr  = WtO + (size_t)(col0 + l16) * D_;
  for (int kk = 0; kk < D_; kk += 32) {
    v16h a0 = load_afrag(ar0, kk, half_);
    v16h a1 = load_afrag(ar1, kk, half_);
#pragma unroll
    for (int ct = 0; ct < 4; ++ct) {
      v16h bf = *(const v16h*)(wr + (size_t)ct * 16 * D_ + kk + 16 * half_);
      acc[0][ct] = wmma_f16(a0, bf, acc[0][ct]);
      acc[1][ct] = wmma_f16(a1, bf, acc[1][ct]);
    }
  }
#pragma unroll
  for (int ct = 0; ct < 4; ++ct) {
    const float bval = bo[col0 + ct * 16 + l16];
#pragma unroll
    for (int rt = 0; rt < 2; ++rt) {
      float* ob = Out + (size_t)(row0 + rt * 16 + 8 * half_) * D_
                      + col0 + ct * 16 + l16;
#pragma unroll
      for (int i = 0; i < 8; ++i) ob[(size_t)i * D_] = acc[rt][ct][i] + bval;
    }
  }
}

// ---------------------------------------------------------------------------
extern "C" void kernel_launch(void* const* d_in, const int* in_sizes, int n_in,
                              void* d_out, int out_size, void* d_ws, size_t ws_size,
                              hipStream_t stream) {
  const float* X  = (const float*)d_in[0];
  const float* Wq = (const float*)d_in[1];
  const float* bq = (const float*)d_in[2];
  const float* Wk = (const float*)d_in[3];
  const float* bk = (const float*)d_in[4];
  const float* Wv = (const float*)d_in[5];
  const float* bv = (const float*)d_in[6];
  const float* Wo = (const float*)d_in[7];
  const float* bo = (const float*)d_in[8];

  h16* ws = (h16*)d_ws;
  const size_t NE = (size_t)B_ * H_ * S_ * DH_;   // 4M elems = 8MB each
  const size_t WE = (size_t)D_ * D_;              // 1M elems = 2MB each
  h16* Qh  = ws;
  h16* Kh  = ws + NE;
  h16* VhT = ws + 2 * NE;
  h16* Ctx = ws + 3 * NE;
  h16* Xh  = ws + 4 * NE;
  h16* WtQ = ws + 5 * NE;
  h16* WtK = WtQ + WE;
  h16* WtV = WtQ + 2 * WE;
  h16* WtO = WtQ + 3 * WE;                        // total 48 MB workspace

  // precision conversion / weight transpose (tiny, memory-bound)
  xcvt_kernel<<<(B_ * S_ * D_) / (256 * 8), 256, 0, stream>>>(X, Xh);
  dim3 gw(D_ / 32, D_ / 32, 4);
  wcvt_kernel<<<gw, 256, 0, stream>>>(Wq, Wk, Wv, Wo, WtQ, WtK, WtV, WtO);

  dim3 g1((B_ * S_) / 32, D_ / 256, 3);           // 4 waves/block over col stripes
  qkv_proj_kernel<<<g1, 128, 0, stream>>>(Xh, WtQ, bq, WtK, bk, WtV, bv,
                                          Qh, Kh, VhT);

  attn_kernel<<<(B_ * H_ * (S_ / 16)) / 8, 256, 0, stream>>>(Qh, Kh, VhT, Ctx);

  dim3 g3((B_ * S_) / 32, D_ / 256, 1);
  out_proj_kernel<<<g3, 128, 0, stream>>>(Ctx, WtO, bo, (float*)d_out);
}
